// FlowMapKoopman_4423816315053
// MI455X (gfx1250) — matrix operand order repaired and checked by reading, not compiled, not run on hardware
//
#include <hip/hip_runtime.h>

// ---------------------------------------------------------------------------
// FlowMapKoopman for MI455X (gfx1250, wave32, WMMA 16x16x32 f16, 320KB LDS/WGP)
// Stage2: M=32 per wave => each 1KB B-fragment LDS read feeds 2 WMMAs (LDS-BW
// roofline fix) and provides 2 independent accumulation chains for the XDL.
// Weight staging uses GLOBAL_LOAD_ASYNC_TO_LDS_B128 (ASYNCcnt) - no VGPR trip.
// ---------------------------------------------------------------------------

typedef __attribute__((ext_vector_type(16))) _Float16 v16h;
typedef __attribute__((ext_vector_type(8)))  float    v8f;

#define ARENA_HALFS 196608          // all weights, f16, fragment-swizzled
#define ARENA_BYTES (ARENA_HALFS * 2)

// Arena segment offsets (halfs). Layer = Kpad x N, frag(s,t) at (s*(N/16)+t)*512.
#define ENC_W1 0        /* 96(68) x 256 */
#define ENC_W2 24576    /* 256 x 256    */
#define ENC_W3 90112    /* 256 x 32     */
#define BIA_W1 98304    /* 32(5) x 64   */
#define BIA_W2 100352   /* 64 x 64      */
#define BIA_W3 104448   /* 64 x 32      */
#define DEC_W1 106496   /* 32 x 256     */
#define DEC_W2 114688   /* 256 x 256    */
#define DEC_W3 180224   /* 256 x 64     */

// ---- fragment address helpers (halfs within one 512-half = 1KB fragment) ----
// A (16x32 f16): lanes 0-15 hold K 0-7 (j0..7), K 16-23 (j8..15); lanes 16-31 K 8-15, 24-31
__device__ __forceinline__ int fragOffA(int kk, int m) {
  int h = (kk >> 3) & 1;
  int j = (kk & 7) | ((kk & 16) >> 1);
  return ((m + (h << 4)) << 4) + j;
}
// B (32x16 f16): lanes 0-15 = cols with K 0-7 (j0..7), K 16-23 (j8..15); lanes 16-31 K 8-15, 24-31
__device__ __forceinline__ int fragOffB(int kk, int n) {
  int h = (kk >> 3) & 1;
  int j = (kk & 7) | ((kk & 16) >> 1);
  return ((n + (h << 4)) << 4) + j;
}

__device__ __forceinline__ v16h ldsFrag(const _Float16* p, int lane) {
  return *(const v16h*)(p + (lane << 4));   // 32B contiguous per lane -> 2x ds_load_b128
}
__device__ __forceinline__ v8f zero8() {
  v8f z = {0.f, 0.f, 0.f, 0.f, 0.f, 0.f, 0.f, 0.f};
  return z;
}
__device__ __forceinline__ v8f wmma_f16(v16h a, v16h b, v8f c) {
  return __builtin_amdgcn_wmma_f32_16x16x32_f16(false, a, false, b, (short)0, c, false, false);
}
// Fast SiLU / sigmoid: v_exp_f32 + v_rcp_f32 (no IEEE div_scale chain).
__device__ __forceinline__ float siluf(float x) {
  return x * __builtin_amdgcn_rcpf(1.f + __expf(-x));
}
__device__ __forceinline__ float sigmf(float x) {
  return __builtin_amdgcn_rcpf(1.f + __expf(-x));
}
__device__ __forceinline__ float clampf(float x, float lo, float hi) {
  return fminf(fmaxf(x, lo), hi);
}

// Async DMA one 16B granule: global -> LDS, tracked by ASYNCcnt (gfx1250).
__device__ __forceinline__ void asyncCopy16(void* ldsDst, const void* gSrc) {
  unsigned           ldsAddr = (unsigned)(unsigned long long)ldsDst;  // addr[31:0] = LDS offset
  unsigned long long gAddr   = (unsigned long long)gSrc;
  asm volatile("global_load_async_to_lds_b128 %0, %1, off"
               :: "v"(ldsAddr), "v"(gAddr) : "memory");
}
__device__ __forceinline__ void asyncWait0() {
  asm volatile("s_wait_asynccnt 0x0" ::: "memory");
}

// Store D-accumulator tile (bias+SiLU) into LDS in A-fragment layout (bias from LDS).
__device__ __forceinline__ void storeTileSiLU(_Float16* H, v8f acc, int t, int lane,
                                              const float* bias) {
  int l16 = lane & 15, hD = lane >> 4;
  int c = t * 16 + l16;                 // D layout: col = lane%16, row = r + 8*(lane/16)
  float bv = bias[c];
  int s = c >> 5, kk = c & 31;
#pragma unroll
  for (int r = 0; r < 8; ++r) {
    float x = siluf(acc[r] + bv);
    H[s * 512 + fragOffA(kk, r + 8 * hD)] = (_Float16)x;
  }
}

// ---------------------------------------------------------------------------
// Kernel 0: f32 -> f16 weight conversion + B-fragment swizzle into d_ws arena.
// ---------------------------------------------------------------------------
__global__ __launch_bounds__(256) void k0_swizzle_weights(
    _Float16* __restrict__ arena,
    const float* __restrict__ eW1, const float* __restrict__ eW2, const float* __restrict__ eW3,
    const float* __restrict__ bW1, const float* __restrict__ bW2, const float* __restrict__ bW3,
    const float* __restrict__ dW1, const float* __restrict__ dW2, const float* __restrict__ dW3) {
  int e = blockIdx.x * blockDim.x + threadIdx.x;
  if (e >= ARENA_HALFS) return;
  const float* W; int Kreal, N, base, local;
  if      (e < 90112)  { if (e < 24576) { W = eW1; Kreal = 68;  N = 256; base = ENC_W1; local = e; }
                         else           { W = eW2; Kreal = 256; N = 256; base = ENC_W2; local = e - ENC_W2; } }
  else if (e < 98304)  { W = eW3; Kreal = 256; N = 32;  base = ENC_W3; local = e - ENC_W3; }
  else if (e < 100352) { W = bW1; Kreal = 5;   N = 64;  base = BIA_W1; local = e - BIA_W1; }
  else if (e < 104448) { W = bW2; Kreal = 64;  N = 64;  base = BIA_W2; local = e - BIA_W2; }
  else if (e < 106496) { W = bW3; Kreal = 64;  N = 32;  base = BIA_W3; local = e - BIA_W3; }
  else if (e < 114688) { W = dW1; Kreal = 32;  N = 256; base = DEC_W1; local = e - DEC_W1; }
  else if (e < 180224) { W = dW2; Kreal = 256; N = 256; base = DEC_W2; local = e - DEC_W2; }
  else                 { W = dW3; Kreal = 256; N = 64;  base = DEC_W3; local = e - DEC_W3; }
  int k = local / N, n = local % N;
  float v = (k < Kreal) ? W[k * N + n] : 0.f;
  int s = k >> 5, kk = k & 31, t = n >> 4;
  arena[base + (s * (N >> 4) + t) * 512 + fragOffB(kk, n & 15)] = (_Float16)v;
}

// ---------------------------------------------------------------------------
// Kernel 1: encoder z = enc([y|g]) via WMMA (96-padded input), plus lambda(g)
// hoisted to per-batch (K-invariant in the reference). 32 WGs x 128 rows.
// LDS halfs: [0,98304) weights | [98304,99392) biases f32 | inF | H  (~289KB)
// ---------------------------------------------------------------------------
__global__ __launch_bounds__(256) void k1_encoder_lambda(
    const float* __restrict__ y, const float* __restrict__ g,
    const float* __restrict__ eb1, const float* __restrict__ eb2, const float* __restrict__ eb3,
    const float* __restrict__ lW1, const float* __restrict__ lb1,
    const float* __restrict__ lW2, const float* __restrict__ lb2,
    const _Float16* __restrict__ arena,
    float* __restrict__ zout, float* __restrict__ lamout) {
  extern __shared__ __align__(128) _Float16 lds[];
  const int tid = threadIdx.x, lane = tid & 31, wave = tid >> 5;
  {   // async-DMA enc weights (fragment-ready) into LDS; biases via VALU (tiny)
    const uint4* src = (const uint4*)arena;
    uint4* dst = (uint4*)lds;
    for (int i = tid; i < 12288; i += 256) asyncCopy16(dst + i, src + i);
    float* bf = (float*)(lds + 98304);
    bf[tid] = eb1[tid];
    bf[256 + tid] = eb2[tid];
    if (tid < 32)  bf[512 + tid] = eb3[tid];
    asyncWait0();
  }
  __syncthreads();
  const float* Leb1 = (const float*)(lds + 98304);
  const float* Leb2 = Leb1 + 256;
  const float* Leb3 = Leb1 + 512;

  const int rbase = blockIdx.x * 128 + wave * 16;
  _Float16* inF = lds + 99392 + wave * 1536;   // 16 x 96 A-fragments
  _Float16* H   = lds + 111680 + wave * 4096;  // 16 x 256 activation staging

  if (lane < 16) {
    int row = rbase + lane;
    for (int c = 0; c < 96; ++c) {
      float v = 0.f;
      if (c < 64)      v = y[row * 64 + c];
      else if (c < 68) v = g[row * 4 + (c - 64)];
      inF[(c >> 5) * 512 + fragOffA(c & 31, lane)] = (_Float16)v;
    }
  }
  // L1: 96 -> 256
  v16h a0 = ldsFrag(inF, lane), a1 = ldsFrag(inF + 512, lane), a2i = ldsFrag(inF + 1024, lane);
  for (int t = 0; t < 16; ++t) {
    v8f acc = zero8();
    acc = wmma_f16(a0,  ldsFrag(lds + (ENC_W1 + (0 * 16 + t) * 512), lane), acc);
    acc = wmma_f16(a1,  ldsFrag(lds + (ENC_W1 + (1 * 16 + t) * 512), lane), acc);
    acc = wmma_f16(a2i, ldsFrag(lds + (ENC_W1 + (2 * 16 + t) * 512), lane), acc);
    storeTileSiLU(H, acc, t, lane, Leb1);
  }
  // L2: 256 -> 256
  v16h af[8];
#pragma unroll
  for (int s = 0; s < 8; ++s) af[s] = ldsFrag(H + s * 512, lane);
  for (int t = 0; t < 16; ++t) {
    v8f acc = zero8();
#pragma unroll
    for (int s = 0; s < 8; ++s)
      acc = wmma_f16(af[s], ldsFrag(lds + (ENC_W2 + (s * 16 + t) * 512), lane), acc);
    storeTileSiLU(H, acc, t, lane, Leb2);
  }
  // L3: 256 -> 32 (z)
#pragma unroll
  for (int s = 0; s < 8; ++s) af[s] = ldsFrag(H + s * 512, lane);
  {
    int l16 = lane & 15, hD = lane >> 4;
    for (int t = 0; t < 2; ++t) {
      v8f acc = zero8();
#pragma unroll
      for (int s = 0; s < 8; ++s)
        acc = wmma_f16(af[s], ldsFrag(lds + (ENC_W3 + (s * 2 + t) * 512), lane), acc);
      int c = t * 16 + l16;
      float bv = Leb3[c];
#pragma unroll
      for (int r = 0; r < 8; ++r)
        zout[(rbase + r + 8 * hD) * 32 + c] = acc[r] + bv;
    }
  }
  // lambda(g): per-batch VALU (hoisted out of the K loop)
  if (tid < 128) {
    int row = blockIdx.x * 128 + tid;
    float gv0 = g[row * 4], gv1 = g[row * 4 + 1], gv2 = g[row * 4 + 2], gv3 = g[row * 4 + 3];
    float hb[64];
#pragma unroll 8
    for (int jh = 0; jh < 64; ++jh) {
      float a = lb1[jh] + gv0 * lW1[jh] + gv1 * lW1[64 + jh]
              + gv2 * lW1[128 + jh] + gv3 * lW1[192 + jh];
      hb[jh] = siluf(a);
    }
#pragma unroll 2
    for (int o = 0; o < 32; ++o) {
      float a = lb2[o];
#pragma unroll 8
      for (int jh = 0; jh < 64; ++jh) a += hb[jh] * lW2[jh * 32 + o];
      a = clampf(a, -10.f, 5.f);
      lamout[row * 32 + o] = clampf(-log1pf(__expf(a)), -10.f, 0.f);
    }
  }
}

// ---------------------------------------------------------------------------
// Kernel 2: per-(b,k): gate (VALU), bias MLP (WMMA), Koopman update, decoder.
// 4 waves x 32 rows (two 16-row A-tiles): every B-fragment read -> 2 WMMAs.
// One chunk = one batch row b (K=128 k's). 1024 blocks x 4 chunks.
// LDS halfs: [0,98304) weights | [98304,99776) biases f32 | [99776,100544)
// scalars | [100544,104640) in/z frags (2/wave) | [104640,137408) H (16KB/wave)
// = 274,816 B.
// ---------------------------------------------------------------------------
__global__ __launch_bounds__(128) void k2_flow_decoder(
    const float* __restrict__ dt, const float* __restrict__ g,
    const float* __restrict__ bb1, const float* __restrict__ bb2, const float* __restrict__ bb3,
    const float* __restrict__ gW1, const float* __restrict__ gb1,
    const float* __restrict__ gW2, const float* __restrict__ gb2,
    const float* __restrict__ db1, const float* __restrict__ db2, const float* __restrict__ db3,
    const _Float16* __restrict__ arena,
    const float* __restrict__ zin, const float* __restrict__ lamin,
    float* __restrict__ out) {
  extern __shared__ __align__(128) _Float16 lds[];
  const int tid = threadIdx.x, lane = tid & 31, wave = tid >> 5;
  {   // async-DMA bias+dec weights (local base 0); bias vectors via VALU (tiny)
    const uint4* src = (const uint4*)(arena + 98304);
    uint4* dst = (uint4*)lds;
    for (int i = tid; i < 12288; i += 128) asyncCopy16(dst + i, src + i);
    float* bf = (float*)(lds + 98304);
    if (tid < 64) bf[tid]       = bb1[tid];
    if (tid < 64) bf[64 + tid]  = bb2[tid];
    if (tid < 32) bf[128 + tid] = bb3[tid];
    for (int i = tid; i < 256; i += 128) bf[160 + i] = db1[i];
    for (int i = tid; i < 256; i += 128) bf[416 + i] = db2[i];
    if (tid < 64) bf[672 + tid] = db3[tid];
    asyncWait0();
  }
  // local weight bases (halfs): biaW1 0, biaW2 2048, biaW3 6144,
  //                             decW1 8192, decW2 16384, decW3 81920
  const float* Lbb1 = (const float*)(lds + 98304);
  const float* Lbb2 = Lbb1 + 64;
  const float* Lbb3 = Lbb1 + 128;
  const float* Ldb1 = Lbb1 + 160;
  const float* Ldb2 = Lbb1 + 416;
  const float* Ldb3 = Lbb1 + 672;
  float* scal   = (float*)(lds + 99776);       // [0,128)=dtn [128,256)=dtp [256,384)=gate
  _Float16* inF = lds + 100544 + wave * 1024;  // two 16x32 frags (in5 -> z_next)
  _Float16* H0  = lds + 104640 + wave * 8192;  // tile0: 16x256 staging
  _Float16* H1  = H0 + 4096;                   // tile1: 16x256 staging
  const int l16 = lane & 15, hD = lane >> 4;

  for (int chunk = blockIdx.x; chunk < 4096; chunk += gridDim.x) {
    const int rowBase = chunk * 128;
    const int b = chunk;                       // rows of a chunk share one batch index
    __syncthreads();                           // weights ready / scal reuse fence
    {
      int row = rowBase + tid;
      float dtn = clampf(dt[row], 0.f, 1.f);
      float dtp = __expf((-6.f + dtn * 8.f) * 2.30258509299404568f); // 10^log10_dt
      dtp = clampf(fmaxf(dtp, 1e-30f), 1e-10f, 100.f);
      float gv0 = g[b * 4], gv1 = g[b * 4 + 1], gv2 = g[b * 4 + 2], gv3 = g[b * 4 + 3];
      float go = gb2[0];
#pragma unroll 8
      for (int jh = 0; jh < 64; ++jh) {
        float a = gb1[jh] + dtn * gW1[jh] + gv0 * gW1[64 + jh] + gv1 * gW1[128 + jh]
                + gv2 * gW1[192 + jh] + gv3 * gW1[256 + jh];
        go += siluf(a) * gW2[jh];
      }
      scal[tid] = dtn; scal[128 + tid] = dtp; scal[256 + tid] = sigmf(go);
    }
    __syncthreads();

    // in5 = [dtn | g | 0-pad]: lanes 0-15 build tile0 rows, lanes 16-31 tile1
    {
      int lr = wave * 32 + hD * 16 + l16;
      float dtn = scal[lr];
#pragma unroll
      for (int c = 0; c < 32; ++c) {
        float v = (c == 0) ? dtn : (c < 5 ? g[b * 4 + (c - 1)] : 0.f);
        inF[hD * 512 + fragOffA(c, l16)] = (_Float16)v;
      }
    }
    // bias MLP L1: 32(5) -> 64
    v16h ain0 = ldsFrag(inF, lane), ain1 = ldsFrag(inF + 512, lane);
    for (int t = 0; t < 4; ++t) {
      v16h Bw = ldsFrag(lds + t * 512, lane);
      v8f acc0 = wmma_f16(ain0, Bw, zero8());
      v8f acc1 = wmma_f16(ain1, Bw, zero8());
      storeTileSiLU(H0, acc0, t, lane, Lbb1);
      storeTileSiLU(H1, acc1, t, lane, Lbb1);
    }
    // bias MLP L2: 64 -> 64
    v16h a20[2], a21[2];
    a20[0] = ldsFrag(H0, lane); a20[1] = ldsFrag(H0 + 512, lane);
    a21[0] = ldsFrag(H1, lane); a21[1] = ldsFrag(H1 + 512, lane);
    for (int t = 0; t < 4; ++t) {
      v8f acc0 = zero8(), acc1 = zero8();
#pragma unroll
      for (int s = 0; s < 2; ++s) {
        v16h Bw = ldsFrag(lds + (2048 + (s * 4 + t) * 512), lane);
        acc0 = wmma_f16(a20[s], Bw, acc0);
        acc1 = wmma_f16(a21[s], Bw, acc1);
      }
      storeTileSiLU(H0, acc0, t, lane, Lbb2);
      storeTileSiLU(H1, acc1, t, lane, Lbb2);
    }
    // bias MLP L3: 64 -> 32, fused Koopman flow-map update -> z_next frags
    a20[0] = ldsFrag(H0, lane); a20[1] = ldsFrag(H0 + 512, lane);
    a21[0] = ldsFrag(H1, lane); a21[1] = ldsFrag(H1 + 512, lane);
    for (int t = 0; t < 2; ++t) {
      v8f acc0 = zero8(), acc1 = zero8();
#pragma unroll
      for (int s = 0; s < 2; ++s) {
        v16h Bw = ldsFrag(lds + (6144 + (s * 2 + t) * 512), lane);
        acc0 = wmma_f16(a20[s], Bw, acc0);
        acc1 = wmma_f16(a21[s], Bw, acc1);
      }
      int zc = t * 16 + l16;
      float bv  = Lbb3[zc];
      float lam = lamin[b * 32 + zc];
      float zv  = zin[b * 32 + zc];
#pragma unroll
      for (int T = 0; T < 2; ++T) {
        v8f acc = T ? acc1 : acc0;
#pragma unroll
        for (int r = 0; r < 8; ++r) {
          int m  = r + 8 * hD;
          int lr = wave * 32 + T * 16 + m;
          float dtp = scal[128 + lr], gate = scal[256 + lr];
          float braw = clampf(acc[r] + bv, -10.f, 10.f);
          float A = __expf(clampf(lam * dtp, -20.f, 0.f));
          float zn = clampf(A * zv + gate * braw * dtp, -100.f, 100.f);
          inF[T * 512 + fragOffA(zc, m)] = (_Float16)zn;
        }
      }
    }
    // decoder L1: 32 -> 256
    v16h az0 = ldsFrag(inF, lane), az1 = ldsFrag(inF + 512, lane);
    for (int t = 0; t < 16; ++t) {
      v16h Bw = ldsFrag(lds + (8192 + t * 512), lane);
      v8f acc0 = wmma_f16(az0, Bw, zero8());
      v8f acc1 = wmma_f16(az1, Bw, zero8());
      storeTileSiLU(H0, acc0, t, lane, Ldb1);
      storeTileSiLU(H1, acc1, t, lane, Ldb1);
    }
    // decoder L2: 256 -> 256 (dominant; B reuse x2, dual acc chains)
    v16h af0[8], af1[8];
#pragma unroll
    for (int s = 0; s < 8; ++s) { af0[s] = ldsFrag(H0 + s * 512, lane);
                                  af1[s] = ldsFrag(H1 + s * 512, lane); }
    for (int t = 0; t < 16; ++t) {
      v8f acc0 = zero8(), acc1 = zero8();
#pragma unroll
      for (int s = 0; s < 8; ++s) {
        v16h Bw = ldsFrag(lds + (16384 + (s * 16 + t) * 512), lane);
        acc0 = wmma_f16(af0[s], Bw, acc0);
        acc1 = wmma_f16(af1[s], Bw, acc1);
      }
      storeTileSiLU(H0, acc0, t, lane, Ldb2);
      storeTileSiLU(H1, acc1, t, lane, Ldb2);
    }
    // decoder L3: 256 -> 64, write output
#pragma unroll
    for (int s = 0; s < 8; ++s) { af0[s] = ldsFrag(H0 + s * 512, lane);
                                  af1[s] = ldsFrag(H1 + s * 512, lane); }
    for (int t = 0; t < 4; ++t) {
      v8f acc0 = zero8(), acc1 = zero8();
#pragma unroll
      for (int s = 0; s < 8; ++s) {
        v16h Bw = ldsFrag(lds + (81920 + (s * 4 + t) * 512), lane);
        acc0 = wmma_f16(af0[s], Bw, acc0);
        acc1 = wmma_f16(af1[s], Bw, acc1);
      }
      int c = t * 16 + l16;
      float bv = Ldb3[c];
#pragma unroll
      for (int r = 0; r < 8; ++r) {
        int m = r + 8 * hD;
        out[(rowBase + wave * 32 + m) * 64 + c]      = acc0[r] + bv;
        out[(rowBase + wave * 32 + 16 + m) * 64 + c] = acc1[r] + bv;
      }
    }
  }
}

// ---------------------------------------------------------------------------
extern "C" void kernel_launch(void* const* d_in, const int* in_sizes, int n_in,
                              void* d_out, int out_size, void* d_ws, size_t ws_size,
                              hipStream_t stream) {
  (void)in_sizes; (void)n_in; (void)out_size; (void)ws_size;
  const float* y    = (const float*)d_in[0];
  const float* dt   = (const float*)d_in[1];
  const float* g    = (const float*)d_in[2];
  const float* eW1  = (const float*)d_in[3];  const float* eb1 = (const float*)d_in[4];
  const float* eW2  = (const float*)d_in[5];  const float* eb2 = (const float*)d_in[6];
  const float* eW3  = (const float*)d_in[7];  const float* eb3 = (const float*)d_in[8];
  const float* lW1  = (const float*)d_in[9];  const float* lb1 = (const float*)d_in[10];
  const float* lW2  = (const float*)d_in[11]; const float* lb2 = (const float*)d_in[12];
  const float* bW1  = (const float*)d_in[13]; const float* bb1 = (const float*)d_in[14];
  const float* bW2  = (const float*)d_in[15]; const float* bb2 = (const float*)d_in[16];
  const float* bW3  = (const float*)d_in[17]; const float* bb3 = (const float*)d_in[18];
  const float* gW1  = (const float*)d_in[19]; const float* gb1 = (const float*)d_in[20];
  const float* gW2  = (const float*)d_in[21]; const float* gb2 = (const float*)d_in[22];
  const float* dW1  = (const float*)d_in[23]; const float* db1 = (const float*)d_in[24];
  const float* dW2  = (const float*)d_in[25]; const float* db2 = (const float*)d_in[26];
  const float* dW3  = (const float*)d_in[27]; const float* db3 = (const float*)d_in[28];

  _Float16* arena = (_Float16*)d_ws;
  float* zws   = (float*)((char*)d_ws + ARENA_BYTES);                  // [4096,32] f32
  float* lamws = (float*)((char*)d_ws + ARENA_BYTES + 4096 * 32 * 4);  // [4096,32] f32

  k0_swizzle_weights<<<ARENA_HALFS / 256, 256, 0, stream>>>(
      arena, eW1, eW2, eW3, bW1, bW2, bW3, dW1, dW2, dW3);

  // 32 WGs x 128 batch rows; dyn-LDS = 144448 halfs = 288,896 B (< 320KB)
  k1_encoder_lambda<<<32, 256, 144448 * 2, stream>>>(
      y, g, eb1, eb2, eb3, lW1, lb1, lW2, lb2, arena, zws, lamws);

  // 1024 persistent blocks x 4 chunks; dyn-LDS = 137408 halfs = 274,816 B
  k2_flow_decoder<<<1024, 128, 137408 * 2, stream>>>(
      dt, g, bb1, bb2, bb3, gW1, gb1, gW2, gb2, db1, db2, db3,
      arena, zws, lamws, (float*)d_out);
}
